// self_attn_60679297958587
// MI455X (gfx1250) — matrix operand order repaired and checked by reading, not compiled, not run on hardware
//
#include <hip/hip_runtime.h>

typedef __attribute__((ext_vector_type(16))) __bf16 v16bf;
typedef __attribute__((ext_vector_type(8)))  __bf16 v8bf;
typedef __attribute__((ext_vector_type(8)))  float  v8f;
typedef __attribute__((ext_vector_type(2)))  float  v2f;

#define BATCH 4
#define CCH   64
#define NPIX  4096
#define CQK   8

__device__ __forceinline__ unsigned short f2bf(float f) {
    union { float f; unsigned u; } v; v.f = f;
    unsigned r = v.u + 0x7FFFu + ((v.u >> 16) & 1u);   // RNE
    return (unsigned short)(r >> 16);
}

// ---------------- Phase 1: fused QKV projection (1x1 convs) ----------------
__global__ void qkv_proj_kernel(const float* __restrict__ x,
                                const float* __restrict__ Wq, const float* __restrict__ bq,
                                const float* __restrict__ Wk, const float* __restrict__ bk,
                                const float* __restrict__ Wv, const float* __restrict__ bv,
                                float* __restrict__ qws, float* __restrict__ kws,
                                unsigned short* __restrict__ vws) {
    __shared__ float w_s[80 * 64];   // rows 0..7 = Wq, 8..15 = Wk, 16..79 = Wv
    __shared__ float b_s[80];
    const int t = threadIdx.x;

    for (int idx = t; idx < 80 * 64; idx += 128) {
        int r = idx >> 6, c = idx & 63;
        float wv;
        if (r < 8)       wv = Wq[r * 64 + c];
        else if (r < 16) wv = Wk[(r - 8) * 64 + c];
        else             wv = Wv[(r - 16) * 64 + c];
        w_s[idx] = wv;
    }
    if (t < 80) b_s[t] = (t < 8) ? bq[t] : ((t < 16) ? bk[t - 8] : bv[t - 16]);
    __syncthreads();

    const int pix = blockIdx.x * 128 + t;     // 0 .. 16383
    const int b = pix >> 12;
    const int n = pix & (NPIX - 1);

    const float* xcol = x + (size_t)b * CCH * NPIX + n;
    float xr[64];
#pragma unroll
    for (int c = 0; c < 64; ++c) xr[c] = xcol[(size_t)c * NPIX];

#pragma unroll
    for (int o = 0; o < 8; ++o) {
        float aq = b_s[o], ak = b_s[8 + o];
#pragma unroll
        for (int c = 0; c < 64; ++c) {
            aq = fmaf(w_s[o * 64 + c],       xr[c], aq);
            ak = fmaf(w_s[(8 + o) * 64 + c], xr[c], ak);
        }
        qws[((size_t)(b * CQK + o)) * NPIX + n] = aq;
        kws[((size_t)(b * CQK + o)) * NPIX + n] = ak;
    }
#pragma unroll 4
    for (int o = 0; o < 64; ++o) {
        float av = b_s[16 + o];
#pragma unroll
        for (int c = 0; c < 64; ++c) av = fmaf(w_s[(16 + o) * 64 + c], xr[c], av);
        vws[((size_t)(b * CCH + o)) * NPIX + n] = f2bf(av);
    }
}

// ---------------- Phase 2: flash attention + output -----------------------
// Block = (batch b, 16 queries). 128 threads = 4 waves.
// Scores: each wave computes a 16x16 strip of S via 2x V_WMMA_F32_16X16X4_F32
//         (A = q fragment, loop-invariant, preloaded in 4 VGPRs).
// Softmax: parallel over all 128 threads (8-segment LDS tree per row).
// PV:     each wave owns 16 channels, 2x V_WMMA_F32_16X16X32_BF16 per chunk.
__global__ void attn_kernel(const float* __restrict__ x,
                            const float* __restrict__ qws,
                            const float* __restrict__ kws,
                            const unsigned short* __restrict__ vws,
                            const float* __restrict__ gamma,
                            float* __restrict__ out) {
    __shared__ float qs[CQK][16];
    __shared__ float S[16][65];                                        // padded
    __shared__ __attribute__((aligned(16))) unsigned short P[16][64];  // bf16 probs
    __shared__ float mrun[16], lrun[16], rfac[16];
    __shared__ float pmax[16][8], psum[16][8];

    const int t  = threadIdx.x;
    const int w  = t >> 5;          // wave id
    const int L  = t & 31;          // lane
    const int m  = L & 15;
    const int hi = L >> 4;

    const int bx = blockIdx.x;
    const int b  = bx >> 8;
    const int i0 = (bx & 255) << 4;

    { int c = t >> 4, i = t & 15;
      qs[c][i] = qws[((size_t)(b * CQK + c)) * NPIX + i0 + i]; }
    if (t < 16) { mrun[t] = -3.0e38f; lrun[t] = 0.0f; }

    v8f acc = {};
    const unsigned short* vrow = vws + ((size_t)(b * CCH + 16 * w + m)) * NPIX;
    const int si  = t >> 3;         // softmax row 0..15
    const int seg = t & 7;          // softmax segment 0..7
    __syncthreads();

    // Loop-invariant A fragments for score WMMA (f32 16x16x4 layout):
    // vgpr0: K = 2*hi, vgpr1: K = 2*hi+1; row M = lane%16.
    v2f aq0, aq1;
    aq0.x = qs[2 * hi][m];     aq0.y = qs[2 * hi + 1][m];
    aq1.x = qs[4 + 2 * hi][m]; aq1.y = qs[5 + 2 * hi][m];

    const float* krow0 = kws + ((size_t)(b * CQK + 2 * hi)) * NPIX;
    const float* krow1 = krow0 + NPIX;
    const float* krow2 = kws + ((size_t)(b * CQK + 4 + 2 * hi)) * NPIX;
    const float* krow3 = krow2 + NPIX;

    for (int j0 = 0; j0 < NPIX; j0 += 64) {
        // ---- scores via f32 WMMA: wave w -> keys [j0+16w, j0+16w+16) ----
        const int jcol = j0 + 16 * w + m;
        v2f bq0, bq1;
        bq0.x = krow0[jcol]; bq0.y = krow1[jcol];
        bq1.x = krow2[jcol]; bq1.y = krow3[jcol];
        v8f sd = {};
        sd = __builtin_amdgcn_wmma_f32_16x16x4_f32(
            false, aq0, false, bq0, (short)0, sd, false, false);
        sd = __builtin_amdgcn_wmma_f32_16x16x4_f32(
            false, aq1, false, bq1, (short)0, sd, false, false);
#pragma unroll
        for (int r = 0; r < 8; ++r) S[r + 8 * hi][16 * w + m] = sd[r];
        __syncthreads();                                   // (1)

        // ---- parallel chunk max: 8 segments per row ----
        {
            float pm = -3.0e38f;
#pragma unroll
            for (int u = 0; u < 8; ++u) pm = fmaxf(pm, S[si][seg * 8 + u]);
            pmax[si][seg] = pm;
        }
        __syncthreads();                                   // (2)
        if (t < 16) {
            float cm = pmax[t][0];
#pragma unroll
            for (int u = 1; u < 8; ++u) cm = fmaxf(cm, pmax[t][u]);
            float mo = mrun[t];
            float mn = fmaxf(mo, cm);
            rfac[t] = __expf(mo - mn);
            mrun[t] = mn;
        }
        __syncthreads();                                   // (3)

        // ---- parallel exp + bf16 probs + partial sums ----
        {
            float mn = mrun[si];
            float ps = 0.0f;
#pragma unroll
            for (int u = 0; u < 8; ++u) {
                float p = __expf(S[si][seg * 8 + u] - mn);
                ps += p;
                P[si][seg * 8 + u] = f2bf(p);
            }
            psum[si][seg] = ps;
        }
        __syncthreads();                                   // (4)
        if (t < 16) {
            float cs = psum[t][0];
#pragma unroll
            for (int u = 1; u < 8; ++u) cs += psum[t][u];
            lrun[t] = lrun[t] * rfac[t] + cs;
        }

        // ---- rescale accumulator (query index i == lane%16 for all vgprs) ----
        float r = rfac[m];
#pragma unroll
        for (int e = 0; e < 8; ++e) acc[e] *= r;

        // ---- PV: two K=32 bf16 WMMAs ----
#pragma unroll
        for (int st = 0; st < 2; ++st) {
            union { v16bf f; v8bf h[2]; } A, Bm;
            const int jb = j0 + 32 * st;
            A.h[0] = *(const v8bf*)(vrow + jb + 8 * hi);
            A.h[1] = *(const v8bf*)(vrow + jb + 16 + 8 * hi);
            Bm.h[0] = *(const v8bf*)(&P[m][32 * st + 8 * hi]);
            Bm.h[1] = *(const v8bf*)(&P[m][32 * st + 16 + 8 * hi]);
            acc = __builtin_amdgcn_wmma_f32_16x16x32_bf16(
                false, A.f, false, Bm.f, (short)0, acc, false, false);
        }
        // no trailing barrier: next-iter barrier (1) orders P reads vs writes
    }
    __syncthreads();

    // ---- epilogue: normalize, gamma*out + x ----
    const float inv = 1.0f / lrun[m];
    const float g   = gamma[0];
#pragma unroll
    for (int e = 0; e < 8; ++e) {
        int c = 16 * w + e + 8 * hi;
        size_t off = ((size_t)(b * CCH + c)) * NPIX + i0 + m;
        out[off] = g * (acc[e] * inv) + x[off];
    }
}

extern "C" void kernel_launch(void* const* d_in, const int* in_sizes, int n_in,
                              void* d_out, int out_size, void* d_ws, size_t ws_size,
                              hipStream_t stream) {
    const float* x     = (const float*)d_in[0];
    const float* Wq    = (const float*)d_in[1];
    const float* bq    = (const float*)d_in[2];
    const float* Wk    = (const float*)d_in[3];
    const float* bk    = (const float*)d_in[4];
    const float* Wv    = (const float*)d_in[5];
    const float* bv    = (const float*)d_in[6];
    const float* gamma = (const float*)d_in[7];
    float* out = (float*)d_out;

    float* qws = (float*)d_ws;
    float* kws = qws + (size_t)BATCH * CQK * NPIX;
    unsigned short* vws = (unsigned short*)(kws + (size_t)BATCH * CQK * NPIX);

    qkv_proj_kernel<<<(BATCH * NPIX) / 128, 128, 0, stream>>>(
        x, Wq, bq, Wk, bk, Wv, bv, qws, kws, vws);

    attn_kernel<<<BATCH * (NPIX / 16), 128, 0, stream>>>(
        x, qws, kws, vws, gamma, out);
}
